// MultiHeadAttention_3238405341672
// MI455X (gfx1250) — matrix operand order repaired and checked by reading, not compiled, not run on hardware
//
#include <hip/hip_runtime.h>

// ---------------------------------------------------------------------------
// MHA forward for MI455X (gfx1250, wave32, WMMA 16x16x32 f16 w/ f32 accum)
// ---------------------------------------------------------------------------

typedef __attribute__((ext_vector_type(4)))  int      v4i;
typedef __attribute__((ext_vector_type(4)))  float    v4f;
typedef __attribute__((ext_vector_type(8)))  float    v8f;
typedef __attribute__((ext_vector_type(8)))  _Float16 v8h;
typedef __attribute__((ext_vector_type(16))) _Float16 v16h;

union H16 { v16h v; v8h h[2]; };

#define WMMA_F16F32(a, b, c) \
  __builtin_amdgcn_wmma_f32_16x16x32_f16(false, (a), false, (b), (short)0, (c), false, false)

__device__ __forceinline__ v8f zero8() {
  v8f z;
#pragma unroll
  for (int i = 0; i < 8; ++i) z[i] = 0.0f;
  return z;
}

// Load 8 consecutive fp32 and convert to 8 packed f16 (two b128 loads + cvt).
__device__ __forceinline__ v8h cvt8(const float* __restrict__ p) {
  v4f a = *(const v4f*)p;
  v4f b = *(const v4f*)(p + 4);
  v8h r;
  r[0] = (_Float16)a[0]; r[1] = (_Float16)a[1];
  r[2] = (_Float16)a[2]; r[3] = (_Float16)a[3];
  r[4] = (_Float16)b[0]; r[5] = (_Float16)b[1];
  r[6] = (_Float16)b[2]; r[7] = (_Float16)b[3];
  return r;
}

// 16-byte async global->LDS copy (per lane). Tracked by ASYNCcnt.
// Probe round 2 revealed the builtin's prototype takes typed v4i pointers:
//   (v4i addrspace(1)* src, v4i addrspace(3)* dst, int imm_offset, int cpol)
#if __has_builtin(__builtin_amdgcn_global_load_async_to_lds_b128)
#define HAVE_ASYNC_LDS 1
#else
#define HAVE_ASYNC_LDS 0
#endif

__device__ __forceinline__ void async_ld16(const _Float16* g, _Float16* l) {
#if HAVE_ASYNC_LDS
  __builtin_amdgcn_global_load_async_to_lds_b128(
      (__attribute__((address_space(1))) v4i*)g,
      (__attribute__((address_space(3))) v4i*)l, 0, 0);
#else
  *(v8h*)l = *(const v8h*)g;
#endif
}

__device__ __forceinline__ void wait_async_lds() {
#if HAVE_ASYNC_LDS
  asm volatile("s_wait_asynccnt 0x0" ::: "memory");
#endif
}

// ---------------------------------------------------------------------------
// GEMM: C[M=8192, N=1024] = A[8192,1024] @ W[1024,1024]^T + bias, then scale.
// Block = 128 threads (4 waves); each wave owns a 16x64 output tile.
// MODE 0: out f16 as [B,H,S,Dh]        (Q with scale=1/8, K with scale=1)
// MODE 2: out f16 as [B,H,Dh,S] (V^T)  -> packed b128 stores
// MODE 3: out f32 row-major [B,S,D]    (final output)
// A_F16: A is f16 (attention output); else fp32 converted in-register.
// ---------------------------------------------------------------------------
template <bool A_F16, int MODE>
__global__ __launch_bounds__(128) void gemm_kernel(
    const void* __restrict__ Aptr, const float* __restrict__ W,
    const float* __restrict__ bias, void* __restrict__ outp, float scale) {
  const int lane = threadIdx.x & 31;
  const int wave = threadIdx.x >> 5;
  const int l15  = lane & 15;
  const int hi   = lane >> 4;
  const int m0   = blockIdx.x * 64 + wave * 16;
  const int n0   = blockIdx.y * 64;

  v8f acc[4];
#pragma unroll
  for (int c = 0; c < 4; ++c) acc[c] = zero8();

  const int arow = m0 + l15;

  for (int k0 = 0; k0 < 1024; k0 += 32) {
    H16 a;
    if constexpr (A_F16) {
      const _Float16* A = (const _Float16*)Aptr;
      a.h[0] = *(const v8h*)(A + (size_t)arow * 1024 + k0 + 8 * hi);
      a.h[1] = *(const v8h*)(A + (size_t)arow * 1024 + k0 + 16 + 8 * hi);
    } else {
      const float* A = (const float*)Aptr;
      a.h[0] = cvt8(A + (size_t)arow * 1024 + k0 + 8 * hi);
      a.h[1] = cvt8(A + (size_t)arow * 1024 + k0 + 16 + 8 * hi);
    }
#pragma unroll
    for (int c = 0; c < 4; ++c) {
      const float* wp = W + (size_t)(n0 + c * 16 + l15) * 1024 + k0 + 8 * hi;
      H16 b;
      b.h[0] = cvt8(wp);
      b.h[1] = cvt8(wp + 16);
      acc[c] = WMMA_F16F32(a.v, b.v, acc[c]);
    }
  }

#pragma unroll
  for (int c = 0; c < 4; ++c) {
    const int n    = n0 + c * 16 + l15;
    const float bv = bias[n];
    if constexpr (MODE == 0) {
      _Float16* out = (_Float16*)outp;
      const int h = n >> 6, dh = n & 63;
#pragma unroll
      for (int r = 0; r < 8; ++r) {
        const int m = m0 + 8 * hi + r;
        const int b = m >> 11, sq = m & 2047;
        out[(((size_t)(b * 16 + h)) * 2048 + sq) * 64 + dh] =
            (_Float16)((acc[c][r] + bv) * scale);
      }
    } else if constexpr (MODE == 2) {
      const int h = n >> 6, dh = n & 63;
      const int b = m0 >> 11;
      const int sq0 = (m0 & 2047) + 8 * hi;  // 8 consecutive rows -> packed store
      v8h pk;
#pragma unroll
      for (int r = 0; r < 8; ++r) pk[r] = (_Float16)((acc[c][r] + bv) * scale);
      *(v8h*)((_Float16*)outp + (((size_t)(b * 16 + h)) * 64 + dh) * 2048 + sq0) = pk;
    } else {  // MODE 3: fp32 row-major
      float* out = (float*)outp;
#pragma unroll
      for (int r = 0; r < 8; ++r) {
        const int m = m0 + 8 * hi + r;
        out[(size_t)m * 1024 + n] = (acc[c][r] + bv) * scale;
      }
    }
  }
}

// ---------------------------------------------------------------------------
// Flash attention, LDS-staged, double-buffered:
//   block = 128 threads (4 waves), all sharing one (b,h) and one key stream.
//   Each wave owns 16 queries (block covers 64). Per 32-key block, the K tile
//   (32x64 f16, 4KB) and V^T tile (64x32 f16, 4KB) are staged into LDS once
//   via GLOBAL_LOAD_ASYNC_TO_LDS_B128 (ASYNCcnt pipeline: copy of tile kb+1
//   overlaps compute of tile kb), then consumed by all 4 waves as WMMA B
//   fragments via ds_load_b128. Online softmax; P restaged wave-locally.
// ---------------------------------------------------------------------------
__global__ __launch_bounds__(128) void attn_kernel(
    const _Float16* __restrict__ Q, const _Float16* __restrict__ K,
    const _Float16* __restrict__ Vt, _Float16* __restrict__ AO) {
  __shared__ __align__(16) _Float16 Kb[2][32][64];   // [buf][key][dh]
  __shared__ __align__(16) _Float16 Vb[2][64][32];   // [buf][dh][key]
  __shared__ __align__(16) _Float16 Pld[4][16][32];  // per-wave P restage

  const int tid  = threadIdx.x;
  const int lane = tid & 31;
  const int wave = tid >> 5;
  const int l15  = lane & 15;
  const int hi   = lane >> 4;
  const int bh   = blockIdx.y;
  const int b    = bh >> 4, h = bh & 15;
  const int q0   = blockIdx.x * 64 + wave * 16;

  const _Float16* Qh = Q + (size_t)bh * 2048 * 64;
  const _Float16* Kh = K + (size_t)bh * 2048 * 64;
  const _Float16* Vh = Vt + (size_t)bh * 64 * 2048;

  // Q A-fragments: qa[kk] covers rows q0.., d in [kk*32, kk*32+32)
  H16 qa[2];
  {
    const _Float16* qp = Qh + (size_t)(q0 + l15) * 64;
    qa[0].h[0] = *(const v8h*)(qp + 8 * hi);
    qa[0].h[1] = *(const v8h*)(qp + 16 + 8 * hi);
    qa[1].h[0] = *(const v8h*)(qp + 32 + 8 * hi);
    qa[1].h[1] = *(const v8h*)(qp + 48 + 8 * hi);
  }

  v8f o[4];
#pragma unroll
  for (int d = 0; d < 4; ++d) o[d] = zero8();
  float mrow[8], srow[8];
#pragma unroll
  for (int r = 0; r < 8; ++r) { mrow[r] = -1e30f; srow[r] = 0.0f; }

  // Stage one 32-key block (K: 256 16B chunks, V: 256 16B chunks; 4/thread).
  auto stage = [&](int kb, int buf) {
    const int kbase = kb * 32;
#pragma unroll
    for (int s = 0; s < 2; ++s) {
      const int c   = tid + s * 128;
      const int row = c >> 3, off = (c & 7) * 8;
      async_ld16(Kh + (size_t)(kbase + row) * 64 + off, &Kb[buf][row][off]);
    }
#pragma unroll
    for (int s = 0; s < 2; ++s) {
      const int c  = tid + s * 128;
      const int dh = c >> 2, off = (c & 3) * 8;
      async_ld16(Vh + (size_t)dh * 2048 + kbase + off, &Vb[buf][dh][off]);
    }
  };

  stage(0, 0);
  wait_async_lds();
  __syncthreads();

  for (int kb = 0; kb < 64; ++kb) {
    const int buf = kb & 1;
    if (kb + 1 < 64) stage(kb + 1, buf ^ 1);  // overlap copy with compute

    // K^T B-fragments from LDS: kf[c][kk], keys c*16.., d in [kk*32, +32)
    H16 kf[2][2];
#pragma unroll
    for (int c = 0; c < 2; ++c) {
      const _Float16* kp = &Kb[buf][c * 16 + l15][0];
      kf[c][0].h[0] = *(const v8h*)(kp + 8 * hi);
      kf[c][0].h[1] = *(const v8h*)(kp + 16 + 8 * hi);
      kf[c][1].h[0] = *(const v8h*)(kp + 32 + 8 * hi);
      kf[c][1].h[1] = *(const v8h*)(kp + 48 + 8 * hi);
    }

    v8f sc[2];
#pragma unroll
    for (int c = 0; c < 2; ++c) {
      v8f z = zero8();
      z = WMMA_F16F32(qa[0].v, kf[c][0].v, z);
      sc[c] = WMMA_F16F32(qa[1].v, kf[c][1].v, z);
    }

    // Online softmax. C-frag row M = r + 8*hi, col N = l15 (+16 for c=1).
#pragma unroll
    for (int r = 0; r < 8; ++r) {
      float vmax = fmaxf(sc[0][r], sc[1][r]);
      vmax = fmaxf(vmax, __shfl_xor(vmax, 8, 32));
      vmax = fmaxf(vmax, __shfl_xor(vmax, 4, 32));
      vmax = fmaxf(vmax, __shfl_xor(vmax, 2, 32));
      vmax = fmaxf(vmax, __shfl_xor(vmax, 1, 32));
      const float nm   = fmaxf(mrow[r], vmax);
      const float corr = __expf(mrow[r] - nm);
      mrow[r] = nm;
      const float p0 = __expf(sc[0][r] - nm);
      const float p1 = __expf(sc[1][r] - nm);
      float ps = p0 + p1;
      ps += __shfl_xor(ps, 8, 32);
      ps += __shfl_xor(ps, 4, 32);
      ps += __shfl_xor(ps, 2, 32);
      ps += __shfl_xor(ps, 1, 32);
      srow[r] = srow[r] * corr + ps;
#pragma unroll
      for (int d = 0; d < 4; ++d) o[d][r] *= corr;
      Pld[wave][8 * hi + r][l15]      = (_Float16)p0;
      Pld[wave][8 * hi + r][16 + l15] = (_Float16)p1;
    }
    // Wave-local LDS exchange: DS ops are in-order per wave; fence the
    // compiler and drain DScnt before re-reading across lanes.
    asm volatile("s_wait_dscnt 0x0" ::: "memory");

    H16 pf;
    pf.h[0] = *(const v8h*)&Pld[wave][l15][8 * hi];
    pf.h[1] = *(const v8h*)&Pld[wave][l15][16 + 8 * hi];

    H16 vf[4];
#pragma unroll
    for (int d = 0; d < 4; ++d) {
      const _Float16* vp = &Vb[buf][d * 16 + l15][0];
      vf[d].h[0] = *(const v8h*)(vp + 8 * hi);
      vf[d].h[1] = *(const v8h*)(vp + 16 + 8 * hi);
    }

#pragma unroll
    for (int d = 0; d < 4; ++d) o[d] = WMMA_F16F32(pf.v, vf[d].v, o[d]);

    // Next tile fully in LDS + all waves done with current buffer.
    wait_async_lds();
    __syncthreads();
  }

  // Normalize and write AO as [B, S, H*Dh] f16 (row-major for final GEMM).
#pragma unroll
  for (int r = 0; r < 8; ++r) {
    const float inv = 1.0f / srow[r];
    const int row = q0 + 8 * hi + r;
#pragma unroll
    for (int d = 0; d < 4; ++d) {
      AO[((size_t)(b * 2048 + row)) * 1024 + h * 64 + d * 16 + l15] =
          (_Float16)(o[d][r] * inv);
    }
  }
}

// ---------------------------------------------------------------------------
extern "C" void kernel_launch(void* const* d_in, const int* in_sizes, int n_in,
                              void* d_out, int out_size, void* d_ws, size_t ws_size,
                              hipStream_t stream) {
  (void)in_sizes; (void)n_in; (void)out_size; (void)ws_size;

  const float* query = (const float*)d_in[0];
  const float* key   = (const float*)d_in[1];
  const float* value = (const float*)d_in[2];
  const float* Wq    = (const float*)d_in[3];
  const float* bq    = (const float*)d_in[4];
  const float* Wk    = (const float*)d_in[5];
  const float* bk    = (const float*)d_in[6];
  const float* Wv    = (const float*)d_in[7];
  const float* bv    = (const float*)d_in[8];
  const float* Wo    = (const float*)d_in[9];
  const float* bo    = (const float*)d_in[10];

  const size_t ELEMS = (size_t)8192 * 1024;
  _Float16* Qbuf = (_Float16*)d_ws;
  _Float16* Kbuf = Qbuf + ELEMS;
  _Float16* Vtb  = Kbuf + ELEMS;
  _Float16* AO   = Vtb + ELEMS;

  dim3 ggrid(128, 16), gblock(128);
  // Q projection pre-scaled by 1/sqrt(Dh)=0.125 (folded into Q so the score
  // WMMAs need no extra multiply).
  gemm_kernel<false, 0><<<ggrid, gblock, 0, stream>>>(query, Wq, bq, Qbuf, 0.125f);
  gemm_kernel<false, 0><<<ggrid, gblock, 0, stream>>>(key,   Wk, bk, Kbuf, 1.0f);
  gemm_kernel<false, 2><<<ggrid, gblock, 0, stream>>>(value, Wv, bv, Vtb,  1.0f);

  attn_kernel<<<dim3(32, 64), 128, 0, stream>>>(Qbuf, Kbuf, Vtb, AO);

  gemm_kernel<true, 3><<<ggrid, gblock, 0, stream>>>(AO, Wo, bo, (float*)d_out, 1.0f);
}
